// RelativeCrossAttention_45380624450282
// MI455X (gfx1250) — compile-verified, matrix operand order checked by reading, not run on hardware
//
#include <hip/hip_runtime.h>

// ---------------------------------------------------------------------------
// RelativeCrossAttention forward for MI455X (gfx1250).
// bf16 WMMA pipeline + Tensor Data Mover (TDM) double-buffered staging.
// ---------------------------------------------------------------------------

typedef __bf16 bf16_t;
typedef __attribute__((ext_vector_type(16))) __bf16 v16bf;
typedef __attribute__((ext_vector_type(8)))  __bf16 v8bf;
typedef __attribute__((ext_vector_type(8)))  float  v8f;
typedef __attribute__((ext_vector_type(4)))  unsigned int u32x4;
typedef __attribute__((ext_vector_type(8)))  int i32x8;
typedef __attribute__((ext_vector_type(4)))  int i32x4;

#define D_MODEL 1024
#define NHEADS  16
#define HDIM    64
#define SEQ     1024
#define BATCH   4
#define MROWS   (BATCH * SEQ)         // 4096
#define RELROWS 2047
#define RELPAD  2112                  // padded rel table rows (zero fill)
#define SCALE_F 0.125f                // 1/sqrt(64)

#if defined(__gfx1250__) && __has_builtin(__builtin_amdgcn_tensor_load_to_lds) && \
    __has_builtin(__builtin_amdgcn_s_wait_tensorcnt)
#define USE_TDM 1
#else
#define USE_TDM 0
#endif

// ---------------- WMMA helpers -------------------------------------------

__device__ inline v8f wmma_bf16(v16bf a, v16bf b, v8f c) {
  return __builtin_amdgcn_wmma_f32_16x16x32_bf16(
      false, a, false, b, (short)0, c, false, false);
}

__device__ inline v8f vzero8() {
  v8f z;
#pragma unroll
  for (int i = 0; i < 8; ++i) z[i] = 0.0f;
  return z;
}

// 16x32 bf16 fragment loader (A layout; identical layout used for B).
// lane 0-15 : M = lane,    K = {0..7, 16..23}
// lane 16-31: M = lane-16, K = {8..15, 24..31}
__device__ inline v16bf load_frag(const bf16_t* tile, int ld) {
  const int lane = threadIdx.x & 31;
  const int m = lane & 15;
  const int h = (lane >> 4) << 3;               // 0 or 8
  const bf16_t* p = tile + (size_t)m * ld;
  v8bf lo = *(const v8bf*)(p + h);
  v8bf hi = *(const v8bf*)(p + 16 + h);
  return __builtin_shufflevector(lo, hi, 0, 1, 2, 3, 4, 5, 6, 7,
                                         8, 9, 10, 11, 12, 13, 14, 15);
}

// ---------------- TDM helpers --------------------------------------------

#if USE_TDM
__device__ inline unsigned lds_addr_of(const void* p) {
  return (unsigned)(unsigned long long)(__attribute__((address_space(3))) const char*)p;
}

// 2D tile DMA: global (row-major, stride_u * 8B between rows) -> LDS,
// data_size = 8 bytes. tile_d0u = row length in 8B units, tile_d1 = rows.
// LDS row pitch handled via D# pad fields: insert (pac+1) DWORDs of pad
// every 2^(pic+1) DWORDs stored.
__device__ inline void tdm_load_2d(unsigned lds_byte, const void* gaddr,
                                   unsigned tile_d0u, unsigned tile_d1,
                                   unsigned stride_u,
                                   unsigned pic, unsigned pac) {
  unsigned long long ga = (unsigned long long)(size_t)gaddr;
  u32x4 g0;
  g0[0] = 1u;                                            // count=1, user mode
  g0[1] = lds_byte;                                      // lds_addr
  g0[2] = (unsigned)ga;                                  // global_addr[31:0]
  g0[3] = (unsigned)((ga >> 32) & 0x1FFFFFFu) | (2u << 30);  // addr[56:32]|type=2
  i32x8 g1;
  g1[0] = (int)((3u << 16) | (1u << 20) | (pic << 22) | (pac << 25)); // 8B, pad_en
  g1[1] = (int)((tile_d0u & 0xFFFFu) << 16);             // tensor_dim0 lo16
  g1[2] = (int)((tile_d0u >> 16) | ((tile_d1 & 0xFFFFu) << 16)); // td0 hi | td1 lo
  g1[3] = (int)((tile_d1 >> 16) | (tile_d0u << 16));     // td1 hi | tile_dim0
  g1[4] = (int)(tile_d1 & 0xFFFFu);                      // tile_dim1 (tile_dim2=0)
  g1[5] = (int)stride_u;                                 // tensor_dim0_stride lo32
  g1[6] = 0;                                             // stride hi / dim1_stride lo
  g1[7] = 0;
  i32x4 z4;
  z4[0] = 0; z4[1] = 0; z4[2] = 0; z4[3] = 0;
  i32x8 z8;
#pragma unroll
  for (int i = 0; i < 8; ++i) z8[i] = 0;
  // 6-arg form (this toolchain): (g0, g1, g2, g3, g4, cpol)
  __builtin_amdgcn_tensor_load_to_lds(g0, g1, z4, z4, z8, 0);
}
#endif

// ---------------- conversion kernels -------------------------------------

__global__ void cvt_kernel(const float* __restrict__ src,
                           bf16_t* __restrict__ dst, int n) {
  int i = blockIdx.x * 256 + threadIdx.x;
  if (i < n) dst[i] = (bf16_t)src[i];
}

__global__ void cvt_pad_kernel(const float* __restrict__ src,
                               bf16_t* __restrict__ dst, int nsrc, int ntot) {
  int i = blockIdx.x * 256 + threadIdx.x;
  if (i < ntot) dst[i] = (i < nsrc) ? (bf16_t)src[i] : (bf16_t)0.0f;
}

// W [K][N] fp32 -> Wt [N][K] bf16 (optionally scaled)
__global__ __launch_bounds__(256) void transpose_cvt_kernel(
    const float* __restrict__ W, bf16_t* __restrict__ Wt, float scale) {
  __shared__ float t[32][33];
  const int n0 = blockIdx.x * 32, k0 = blockIdx.y * 32;
  const int tx = threadIdx.x & 31, ty = threadIdx.x >> 5;  // 32 x 8
#pragma unroll
  for (int r = 0; r < 32; r += 8)
    t[ty + r][tx] = W[(size_t)(k0 + ty + r) * D_MODEL + n0 + tx];
  __syncthreads();
#pragma unroll
  for (int r = 0; r < 32; r += 8)
    Wt[(size_t)(n0 + ty + r) * D_MODEL + k0 + tx] = (bf16_t)(t[tx][ty + r] * scale);
}

// ---------------- GEMM: C = A(bf16) @ Wt^T + bias ------------------------
// A: M x 1024 row-major bf16.  Bt: 1024 x 1024 bf16 ([N][K]).
// mode 0: bf16 out, per-head natural  [b*16+h][s][d]   (Q, K)
// mode 2: bf16 out, per-head transposed [b*16+h][d][s] (V)
// mode 3: fp32 out, plain [row][col]                   (final projection)

__global__ __launch_bounds__(256) void gemm_bf16(
    const bf16_t* __restrict__ A, const bf16_t* __restrict__ Bt,
    const float* __restrict__ bias, float bias_scale,
    void* __restrict__ Cout, int mode) {
  __shared__ __align__(16) bf16_t As[2][128][40];
  __shared__ __align__(16) bf16_t Bs[2][128][40];

  const int tid = threadIdx.x;
  const int wave = tid >> 5, lane = tid & 31;
  const int lm = lane & 15, lh = lane >> 4;
  const int m0 = blockIdx.y * 128;
  const int n0 = blockIdx.x * 128;
  const int mhalf = (wave >> 2) * 64;   // 0 or 64
  const int nquad = (wave & 3) * 32;    // 0..96

  v8f acc[4][2];
#pragma unroll
  for (int i = 0; i < 4; ++i)
#pragma unroll
    for (int j = 0; j < 2; ++j) acc[i][j] = vzero8();

#if USE_TDM
  // prologue: DMA first k-tile into buffer 0 (A:128x32, B:128x32, pitch 80B)
  if (wave == 0) {
    tdm_load_2d(lds_addr_of(&As[0][0][0]), A + (size_t)m0 * D_MODEL,
                8, 128, 256, /*pad every 16 dw*/3, /*pad 4 dw*/3);
    tdm_load_2d(lds_addr_of(&Bs[0][0][0]), Bt + (size_t)n0 * D_MODEL,
                8, 128, 256, 3, 3);
    __builtin_amdgcn_s_wait_tensorcnt(0);
  }
  __syncthreads();
#endif

  int cur = 0;
  for (int k0 = 0; k0 < D_MODEL; k0 += 32) {
#if USE_TDM
    if (wave == 0 && k0 + 32 < D_MODEL) {   // prefetch next k-tile
      tdm_load_2d(lds_addr_of(&As[cur ^ 1][0][0]),
                  A + (size_t)m0 * D_MODEL + k0 + 32, 8, 128, 256, 3, 3);
      tdm_load_2d(lds_addr_of(&Bs[cur ^ 1][0][0]),
                  Bt + (size_t)n0 * D_MODEL + k0 + 32, 8, 128, 256, 3, 3);
    }
#else
#pragma unroll
    for (int i = 0; i < 2; ++i) {
      int c = tid * 2 + i;                 // 0..511
      int row = c >> 2, cb = (c & 3) * 8;
      *(v8bf*)(&As[cur][row][cb]) =
          *(const v8bf*)(A + (size_t)(m0 + row) * D_MODEL + k0 + cb);
      *(v8bf*)(&Bs[cur][row][cb]) =
          *(const v8bf*)(Bt + (size_t)(n0 + row) * D_MODEL + k0 + cb);
    }
    __syncthreads();
#endif

    v16bf bfrag[2];
#pragma unroll
    for (int nt = 0; nt < 2; ++nt)
      bfrag[nt] = load_frag(&Bs[cur][nquad + nt * 16][0], 40);
#pragma unroll
    for (int mt = 0; mt < 4; ++mt) {
      v16bf afrag = load_frag(&As[cur][mhalf + mt * 16][0], 40);
#pragma unroll
      for (int nt = 0; nt < 2; ++nt)
        acc[mt][nt] = wmma_bf16(afrag, bfrag[nt], acc[mt][nt]);
    }

#if USE_TDM
    if (wave == 0) __builtin_amdgcn_s_wait_tensorcnt(0);
    __syncthreads();
    cur ^= 1;
#else
    __syncthreads();
#endif
  }

#pragma unroll
  for (int mt = 0; mt < 4; ++mt)
#pragma unroll
    for (int nt = 0; nt < 2; ++nt)
#pragma unroll
      for (int r = 0; r < 8; ++r) {
        int row = m0 + mhalf + mt * 16 + lh * 8 + r;
        int col = n0 + nquad + nt * 16 + lm;
        float v = acc[mt][nt][r] + bias[col] * bias_scale;
        if (mode == 3) {
          ((float*)Cout)[(size_t)row * D_MODEL + col] = v;
        } else {
          int b = row >> 10, s = row & 1023;
          int hh = col >> 6, d = col & 63;
          size_t bh = (size_t)(b * NHEADS + hh);
          bf16_t bv = (bf16_t)v;
          if (mode == 2)
            ((bf16_t*)Cout)[(bh * HDIM + d) * SEQ + s] = bv;    // V transposed
          else
            ((bf16_t*)Cout)[(bh * SEQ + s) * HDIM + d] = bv;    // Q/K natural
        }
      }
}

// ---------------- fused flash attention with relative bias ---------------
// grid: 1024 blocks = (b*16+h) * 16 i-blocks ; 256 threads = 8 waves.
// LDS: Q(64x72) K[2](64x72) V[2](64x72 as [d][j]) P(64x72) R(64x1160) + stats.

#define TILE_E (64 * 72)
#define QS_OFF 0
#define KS_OFF TILE_E
#define VS_OFF (3 * TILE_E)
#define PS_OFF (5 * TILE_E)
#define RS_OFF (6 * TILE_E)
#define RLD    1160
#define STAT_BYTE_OFF ((RS_OFF + 64 * RLD) * 2)
#define SMEM_ATTN (STAT_BYTE_OFF + 7 * 64 * 4)

__global__ __launch_bounds__(256) void attn_kernel(
    const bf16_t* __restrict__ Qh, const bf16_t* __restrict__ Kh,
    const bf16_t* __restrict__ Vt, const bf16_t* __restrict__ Tbf,
    bf16_t* __restrict__ AO) {
  extern __shared__ __align__(16) char smem_raw[];
  bf16_t* Qs = (bf16_t*)smem_raw + QS_OFF;
  bf16_t* Ks = (bf16_t*)smem_raw + KS_OFF;   // 2 buffers
  bf16_t* Vs = (bf16_t*)smem_raw + VS_OFF;   // 2 buffers
  bf16_t* Ps = (bf16_t*)smem_raw + PS_OFF;
  bf16_t* Rs = (bf16_t*)smem_raw + RS_OFF;
  float* stat_m     = (float*)(smem_raw + STAT_BYTE_OFF);
  float* stat_l     = stat_m + 64;
  float* stat_alpha = stat_m + 128;
  float* partmax    = stat_m + 192;   // [2][64]
  float* partsum    = stat_m + 320;   // [2][64]

  const int tid = threadIdx.x, wave = tid >> 5, lane = tid & 31;
  const int lm = lane & 15, lh = lane >> 4;
  const int ib = blockIdx.x & 15;
  const int bh = blockIdx.x >> 4;          // b*16 + h
  const int i0 = ib * 64;
  const int bb = bh >> 4, hh = bh & 15;

  const bf16_t* Qg = Qh + ((size_t)bh * SEQ + i0) * HDIM;
  const bf16_t* Kg = Kh + (size_t)bh * SEQ * HDIM;
  const bf16_t* Vg = Vt + (size_t)bh * HDIM * SEQ;

  // ---- stage Q block (64x64) + first K/V tiles ----
#if USE_TDM
  if (wave == 0) {
    tdm_load_2d(lds_addr_of(Qs), Qg, 16, 64, 16, /*pad every 32 dw*/4, /*4 dw*/3);
    tdm_load_2d(lds_addr_of(Ks), Kg, 16, 64, 16, 4, 3);
    tdm_load_2d(lds_addr_of(Vs), Vg, 16, 64, 256, 4, 3);
    __builtin_amdgcn_s_wait_tensorcnt(0);
  }
  __syncthreads();
#else
#pragma unroll
  for (int i = 0; i < 2; ++i) {
    int c = tid * 2 + i;                   // 512 chunks of 8 bf16
    int row = c >> 3, cb = (c & 7) * 8;
    *(v8bf*)(Qs + row * 72 + cb) = *(const v8bf*)(Qg + (size_t)row * HDIM + cb);
  }
  __syncthreads();
#endif

  // ---- R = Q_blk @ T[i0 : i0+1152]^T  (skewed rel-bias table) ----
  const int Rmt = (wave & 3) * 16;
  const int Rnh = (wave >> 2) * 64;
  {
    v16bf aq0 = load_frag(Qs + Rmt * 72, 72);
    v16bf aq1 = load_frag(Qs + Rmt * 72 + 32, 72);
    for (int c = 0; c < 9; ++c) {
      const int ncol0 = c * 128 + Rnh;
      __builtin_prefetch(Tbf + (size_t)(i0 + ncol0 + 128) * HDIM, 0, 0);
      v8f racc[4];
#pragma unroll
      for (int nt = 0; nt < 4; ++nt) racc[nt] = vzero8();
#pragma unroll
      for (int nt = 0; nt < 4; ++nt) {
        const bf16_t* tb = Tbf + (size_t)(i0 + ncol0 + nt * 16) * HDIM;
        racc[nt] = wmma_bf16(aq0, load_frag(tb, HDIM), racc[nt]);
        racc[nt] = wmma_bf16(aq1, load_frag(tb + 32, HDIM), racc[nt]);
      }
#pragma unroll
      for (int nt = 0; nt < 4; ++nt)
#pragma unroll
        for (int r = 0; r < 8; ++r) {
          int row = Rmt + lh * 8 + r, col = ncol0 + nt * 16 + lm;
          Rs[(size_t)row * RLD + col] = (bf16_t)racc[nt][r];
        }
    }
  }
  if (tid < 64) { stat_m[tid] = -1e30f; stat_l[tid] = 0.0f; }
  __syncthreads();

  // ---- flash loop over KV ----
  const int Smt = (wave & 3) * 16;     // this wave's 16 query rows
  const int Snh = (wave >> 2) * 32;    // this wave's 32 columns
  v8f oacc[2] = {vzero8(), vzero8()};

  int cur = 0;
  for (int j0 = 0; j0 < SEQ; j0 += 64) {
#if USE_TDM
    if (wave == 0 && j0 + 64 < SEQ) {  // prefetch next K/V tiles
      tdm_load_2d(lds_addr_of(Ks + (cur ^ 1) * TILE_E), Kg + (size_t)(j0 + 64) * HDIM,
                  16, 64, 16, 4, 3);
      tdm_load_2d(lds_addr_of(Vs + (cur ^ 1) * TILE_E), Vg + (j0 + 64),
                  16, 64, 256, 4, 3);
    }
#else
#pragma unroll
    for (int i = 0; i < 2; ++i) {
      int c = tid * 2 + i;
      int row = c >> 3, cb = (c & 7) * 8;
      *(v8bf*)(Ks + row * 72 + cb) =
          *(const v8bf*)(Kg + (size_t)(j0 + row) * HDIM + cb);
      *(v8bf*)(Vs + row * 72 + cb) =
          *(const v8bf*)(Vg + (size_t)row * SEQ + j0 + cb);
    }
    __syncthreads();
#endif
    const bf16_t* Ksc = Ks + cur * TILE_E;
    const bf16_t* Vsc = Vs + cur * TILE_E;

    // S = Q @ K'^T  (K pre-scaled by 1/sqrt(d))
    v8f s[2] = {vzero8(), vzero8()};
    v16bf a0 = load_frag(Qs + Smt * 72, 72);
    v16bf a1 = load_frag(Qs + Smt * 72 + 32, 72);
#pragma unroll
    for (int nt = 0; nt < 2; ++nt) {
      const bf16_t* kb = Ksc + (Snh + nt * 16) * 72;
      s[nt] = wmma_bf16(a0, load_frag(kb, 72), s[nt]);
      s[nt] = wmma_bf16(a1, load_frag(kb + 32, 72), s[nt]);
    }
    // + relative bias: skewed gather from R
#pragma unroll
    for (int nt = 0; nt < 2; ++nt)
#pragma unroll
      for (int r = 0; r < 8; ++r) {
        int ii = Smt + lh * 8 + r;
        int jj = Snh + nt * 16 + lm;
        s[nt][r] += (float)Rs[(size_t)ii * RLD + (ii - j0 - jj + 1023)];
      }

    // per-row max across this wave's 32 cols (16-lane shuffle reduce)
    float pm[8];
#pragma unroll
    for (int r = 0; r < 8; ++r) {
      float v = fmaxf(s[0][r], s[1][r]);
#pragma unroll
      for (int off = 1; off < 16; off <<= 1) v = fmaxf(v, __shfl_xor(v, off, 32));
      pm[r] = v;
    }
    if (lm == 0)
#pragma unroll
      for (int r = 0; r < 8; ++r)
        partmax[(wave >> 2) * 64 + Smt + lh * 8 + r] = pm[r];
    __syncthreads();
    if (tid < 64) {
      float mo = stat_m[tid];
      float mn = fmaxf(mo, fmaxf(partmax[tid], partmax[64 + tid]));
      stat_m[tid] = mn;
      stat_alpha[tid] = __expf(mo - mn);
    }
    __syncthreads();

    // P = exp(S - m), rescale O accumulators, stage P, row-sum
    float ps[8];
#pragma unroll
    for (int r = 0; r < 8; ++r) {
      int ii = Smt + lh * 8 + r;
      float mn = stat_m[ii];
      float al = stat_alpha[ii];
      oacc[0][r] *= al;
      oacc[1][r] *= al;
      float p0 = __expf(s[0][r] - mn);
      float p1 = __expf(s[1][r] - mn);
      Ps[ii * 72 + Snh + lm] = (bf16_t)p0;
      Ps[ii * 72 + Snh + 16 + lm] = (bf16_t)p1;
      float v = p0 + p1;
#pragma unroll
      for (int off = 1; off < 16; off <<= 1) v += __shfl_xor(v, off, 32);
      ps[r] = v;
    }
    if (lm == 0)
#pragma unroll
      for (int r = 0; r < 8; ++r)
        partsum[(wave >> 2) * 64 + Smt + lh * 8 + r] = ps[r];
    __syncthreads();
    if (tid < 64)
      stat_l[tid] = stat_l[tid] * stat_alpha[tid] + partsum[tid] + partsum[64 + tid];

    // O += P @ V  (V staged as [d][j] so B-fragment is contiguous in j)
    v16bf pa0 = load_frag(Ps + Smt * 72, 72);
    v16bf pa1 = load_frag(Ps + Smt * 72 + 32, 72);
#pragma unroll
    for (int nt = 0; nt < 2; ++nt) {
      const bf16_t* vb = Vsc + (Snh + nt * 16) * 72;
      oacc[nt] = wmma_bf16(pa0, load_frag(vb, 72), oacc[nt]);
      oacc[nt] = wmma_bf16(pa1, load_frag(vb + 32, 72), oacc[nt]);
    }

#if USE_TDM
    if (wave == 0) __builtin_amdgcn_s_wait_tensorcnt(0);
    __syncthreads();
    cur ^= 1;
#else
    __syncthreads();
#endif
  }

  // ---- normalize and write [b][s][h*64+d] ----
#pragma unroll
  for (int nt = 0; nt < 2; ++nt)
#pragma unroll
    for (int r = 0; r < 8; ++r) {
      int ii = Smt + lh * 8 + r;
      int d = Snh + nt * 16 + lm;
      float v = oacc[nt][r] / stat_l[ii];
      AO[((size_t)bb * SEQ + i0 + ii) * D_MODEL + hh * HDIM + d] = (bf16_t)v;
    }
}

// ---------------- host orchestration -------------------------------------

extern "C" void kernel_launch(void* const* d_in, const int* in_sizes, int n_in,
                              void* d_out, int out_size, void* d_ws, size_t ws_size,
                              hipStream_t stream) {
  const float* query = (const float*)d_in[0];
  const float* keyv  = (const float*)d_in[1];
  const float* Wq = (const float*)d_in[2];
  const float* bq = (const float*)d_in[3];
  const float* Wk = (const float*)d_in[4];
  const float* bk = (const float*)d_in[5];
  const float* Wv = (const float*)d_in[6];
  const float* bv = (const float*)d_in[7];
  const float* Wo = (const float*)d_in[8];
  const float* bo = (const float*)d_in[9];
  const float* relT = (const float*)d_in[10];

  char* ws = (char*)d_ws;
  size_t off = 0;
  auto alloc = [&](size_t bytes) -> void* {
    off = (off + 255) & ~(size_t)255;
    void* p = ws + off;
    off += bytes;
    return p;
  };

  const size_t NE = (size_t)MROWS * D_MODEL;  // 4M elements
  bf16_t* Aq  = (bf16_t*)alloc(NE * 2);
  bf16_t* Akv = (bf16_t*)alloc(NE * 2);
  bf16_t* WtQ = (bf16_t*)alloc((size_t)D_MODEL * D_MODEL * 2);
  bf16_t* WtK = (bf16_t*)alloc((size_t)D_MODEL * D_MODEL * 2);
  bf16_t* WtV = (bf16_t*)alloc((size_t)D_MODEL * D_MODEL * 2);
  bf16_t* WtO = (bf16_t*)alloc((size_t)D_MODEL * D_MODEL * 2);
  bf16_t* Tbf = (bf16_t*)alloc((size_t)RELPAD * HDIM * 2);
  bf16_t* Qh  = (bf16_t*)alloc(NE * 2);
  bf16_t* Kh  = (bf16_t*)alloc(NE * 2);
  bf16_t* Vt  = (bf16_t*)alloc(NE * 2);
  bf16_t* AO  = (bf16_t*)alloc(NE * 2);

  (void)hipFuncSetAttribute((const void*)attn_kernel,
                            hipFuncAttributeMaxDynamicSharedMemorySize, SMEM_ATTN);

  // conversions
  cvt_kernel<<<(int)(NE / 256), 256, 0, stream>>>(query, Aq, (int)NE);
  cvt_kernel<<<(int)(NE / 256), 256, 0, stream>>>(keyv, Akv, (int)NE);
  cvt_pad_kernel<<<(RELPAD * HDIM + 255) / 256, 256, 0, stream>>>(
      relT, Tbf, RELROWS * HDIM, RELPAD * HDIM);
  transpose_cvt_kernel<<<dim3(32, 32), 256, 0, stream>>>(Wq, WtQ, 1.0f);
  transpose_cvt_kernel<<<dim3(32, 32), 256, 0, stream>>>(Wk, WtK, SCALE_F);
  transpose_cvt_kernel<<<dim3(32, 32), 256, 0, stream>>>(Wv, WtV, 1.0f);
  transpose_cvt_kernel<<<dim3(32, 32), 256, 0, stream>>>(Wo, WtO, 1.0f);

  // projections (K pre-scaled, V transposed per head)
  dim3 ggrid(8, 32);
  gemm_bf16<<<ggrid, 256, 0, stream>>>(Aq,  WtQ, bq, 1.0f,    Qh, 0);
  gemm_bf16<<<ggrid, 256, 0, stream>>>(Akv, WtK, bk, SCALE_F, Kh, 0);
  gemm_bf16<<<ggrid, 256, 0, stream>>>(Akv, WtV, bv, 1.0f,    Vt, 2);

  // fused attention with relative position bias
  attn_kernel<<<1024, 256, SMEM_ATTN, stream>>>(Qh, Kh, Vt, Tbf, AO);

  // output projection -> fp32 result
  gemm_bf16<<<ggrid, 256, 0, stream>>>(AO, WtO, bo, 1.0f, d_out, 3);
}
